// MHSelfAttention_5918464934709
// MI455X (gfx1250) — matrix operand order repaired and checked by reading, not compiled, not run on hardware
//
#include <hip/hip_runtime.h>

#define DEVINL __device__ __forceinline__

typedef __attribute__((ext_vector_type(16))) _Float16 v16h;
typedef __attribute__((ext_vector_type(8)))  _Float16 v8h;
typedef __attribute__((ext_vector_type(8)))  float    v8f;

// Problem constants (reference: B,T,C = 2,1536,1536; H,QK,V = 8,64,192; F=192)
constexpr int Bc  = 2;
constexpr int Tc  = 1536;
constexpr int Cc  = 1536;
constexpr int Hc  = 8;
constexpr int QKc = 64;
constexpr int Vc  = 192;
constexpr int Fc  = 192;
constexpr int Rc  = 2 * Tc - 1;   // 3071 relative positions
constexpr int Rp  = 2 * Tc;       // padded to 3072 rows

DEVINL v8f vzero8() {
  v8f z;
#pragma unroll
  for (int i = 0; i < 8; ++i) z[i] = 0.f;
  return z;
}

DEVINL v8f wmma_f16(v16h a, v16h b, v8f c) {
  // D = A(16x32 f16) x B(32x16 f16) + C(16x16 f32)
  return __builtin_amdgcn_wmma_f32_16x16x32_f16(false, a, false, b, (short)0, c, false, false);
}

union FragU { v16h v; v8h h[2]; };

// A fragment 16x32 f16. Lane: m = lane&15. K halves: lanes<16 -> {0..7,16..23},
// lanes>=16 -> {8..15,24..31} (ISA 7.12.2 16-bit A layout). rowptr = &A[m][0].
DEVINL v16h load_fragA(const _Float16* rowptr, int hi) {
  FragU f;
  const int c0 = hi * 8;
  f.h[0] = *(const v8h*)(rowptr + c0);
  f.h[1] = *(const v8h*)(rowptr + c0 + 16);
  return f.v;
}

// B fragment 32x16 f16. Lane: n = lane&15. K: lanes<16 -> {0..7,16..23}, lanes>=16 -> {8..15,24..31}
// (matches A layout). rowptr points at the 32 contiguous K values of col n.
DEVINL v16h load_fragB(const _Float16* rowptr, int hi) {
  FragU f;
  const int c0 = hi * 8;
  f.h[0] = *(const v8h*)(rowptr + c0);
  f.h[1] = *(const v8h*)(rowptr + c0 + 16);
  return f.v;
}

// ---------------------------------------------------------------------------
// f32 -> f16 conversion kernels
// ---------------------------------------------------------------------------
__global__ void cvt_f32_to_f16(const float* __restrict__ in, _Float16* __restrict__ out, int n) {
  int i = blockIdx.x * 256 + threadIdx.x;
  if (i < n) out[i] = (_Float16)in[i];
}

// W[K][N] f32 -> WT[N][K] f16 (so B fragments load contiguous K runs)
__global__ void cvt_transpose_f16(const float* __restrict__ in, _Float16* __restrict__ out,
                                  int K, int N) {
  int i = blockIdx.x * 256 + threadIdx.x;
  if (i < K * N) {
    int k = i / N, n = i % N;
    out[(size_t)n * K + k] = (_Float16)in[i];
  }
}

// ---------------------------------------------------------------------------
// Positional features: one wave per position p (0..3071), lane = basis index (nb=32)
// Row 3071 is zero padding for the rel_k GEMM.
// ---------------------------------------------------------------------------
__global__ __launch_bounds__(128) void posfeat_kernel(_Float16* __restrict__ pf) {
  const int w = threadIdx.x >> 5;
  const int i = threadIdx.x & 31;          // basis index, nb == 32
  const int p = blockIdx.x * 4 + w;
  _Float16* row = pf + (size_t)p * Fc;
  if (p >= Rc) {                            // zero padding row
    for (int j = i; j < Fc; j += 32) row[j] = (_Float16)0.f;
    return;
  }
  const float pos = (float)(p - (Tc - 1));
  const float ap = fabsf(pos);
  const float sgn = (pos > 0.f) ? 1.f : ((pos < 0.f) ? -1.f : 0.f);
  // exponential basis: half_life = 2^linspace(3, log2(T), 32); f = 2^(-ap/hl)
  const float maxr = log2f((float)Tc);
  const float hl = exp2f(3.f + (float)i * (maxr - 3.f) / 31.f);
  const float fe = exp2f(-ap / hl);
  // central-mask basis
  const float cw = exp2f((float)(i + 1)) - 1.f;
  const float fc = (cw > ap) ? 1.f : 0.f;
  // gamma basis
  const float stddev = (float)Tc / 64.f;                 // T/(2*nb) = 24
  const float mean = ((float)Tc / 32.f) +
                     (float)i * ((float)Tc - (float)Tc / 32.f) / 31.f;
  const float conc = (mean / stddev) * (mean / stddev);
  const float rate = mean / (stddev * stddev);
  const float lnorm = lgammaf(conc) - conc * logf(rate);
  const float lu = (ap > 0.f) ? ((conc - 1.f) * logf(ap) - rate * ap) : -1e30f;
  const float prob = expf(lu - lnorm) + 1e-8f;
  float mx = prob;
#pragma unroll
  for (int d = 1; d < 32; d <<= 1) mx = fmaxf(mx, __shfl_xor(mx, d, 32));
  const float fg = prob / mx;
  row[i]            = (_Float16)fe;
  row[32 + i]       = (_Float16)fc;
  row[64 + i]       = (_Float16)fg;
  row[96 + i]       = (_Float16)(sgn * fe);
  row[128 + i]      = (_Float16)(sgn * fc);
  row[160 + i]      = (_Float16)(sgn * fg);
}

// ---------------------------------------------------------------------------
// Generic WMMA GEMM: C[M,N] = A[M,K] @ BT[N,K]^T, f32 accumulate.
// Block = 4 waves; wave tile = 16x64; block tile = 64x64.
// Loads for a K-step are batched ahead of the 4 WMMAs so the scheduler can
// clause them and use partial s_wait_loadcnt values (loads return in order).
// ---------------------------------------------------------------------------
template <int MODE>
__global__ __launch_bounds__(128) void gemm_f16_wmma(
    const _Float16* __restrict__ A, const _Float16* __restrict__ BT,
    int M, int N, int K,
    float* __restrict__ fout, _Float16* __restrict__ hout0, _Float16* __restrict__ hout1,
    const float* __restrict__ bias0, const float* __restrict__ bias1) {
  const int w = threadIdx.x >> 5, lane = threadIdx.x & 31;
  const int lm = lane & 15, hi = lane >> 4;
  const int m0 = blockIdx.y * 64 + w * 16;
  const int n0 = blockIdx.x * 64;
  v8f acc[4];
#pragma unroll
  for (int j = 0; j < 4; ++j) acc[j] = vzero8();
  const _Float16* arow = A + (size_t)(m0 + lm) * K;
  const _Float16* brow = BT + (size_t)(n0 + lm) * K;
#pragma unroll 2
  for (int kk = 0; kk < K; kk += 32) {
    const v16h aF = load_fragA(arow + kk, hi);
    v16h bF[4];
#pragma unroll
    for (int j = 0; j < 4; ++j)
      bF[j] = load_fragB(brow + (size_t)(16 * j) * K + kk, hi);
#pragma unroll
    for (int j = 0; j < 4; ++j)
      acc[j] = wmma_f16(aF, bF[j], acc[j]);
  }
#pragma unroll
  for (int j = 0; j < 4; ++j) {
#pragma unroll
    for (int g = 0; g < 8; ++g) {
      const int m = m0 + g + 8 * hi;           // C layout: lane n = lm, row = g + 8*hi
      const int n = n0 + 16 * j + lm;
      const float val = acc[j][g];
      if constexpr (MODE == 0) {               // final projection, f32 + bias -> d_out
        fout[(size_t)m * N + n] = val + bias0[n];
      } else if constexpr (MODE == 1) {        // Q: scale then + r_w_bias / r_r_bias -> [B,H,T,QK]
        const int b = m / Tc, t = m % Tc, h = n >> 6, d = n & 63;
        const size_t o = ((size_t)(b * Hc + h) * Tc + t) * QKc + d;
        const float s = val * 0.125f;          // QK^-0.5
        hout0[o] = (_Float16)(s + bias0[n]);
        hout1[o] = (_Float16)(s + bias1[n]);
      } else if constexpr (MODE == 2) {        // K -> [B,H,T,QK]
        const int b = m / Tc, t = m % Tc, h = n >> 6, d = n & 63;
        hout0[((size_t)(b * Hc + h) * Tc + t) * QKc + d] = (_Float16)val;
      } else if constexpr (MODE == 3) {        // V -> transposed [B,H,V,T]
        const int b = m / Tc, t = m % Tc, h = n / Vc, vd = n % Vc;
        hout0[((size_t)(b * Hc + h) * Vc + vd) * Tc + t] = (_Float16)val;
      } else {                                 // MODE 4: rel_k -> [H, Rp, QK]
        const int h = n >> 6, d = n & 63;
        hout0[((size_t)h * Rp + m) * QKc + d] = (_Float16)val;
      }
    }
  }
}

// ---------------------------------------------------------------------------
// Fused flash attention with relative-shift folded into a banded WMMA.
// Block = 4 waves; each wave owns a 16-row t-tile; grid = (T/64, B*H).
// logits[t,s] = (qw_t . k_s) + (qr_t . rel_k[s - t + T - 1])
// Per 32-col s-chunk: content (4 WMMAs) + 16x48 rel band (6 WMMAs) +
// diagonal extraction via ds_bpermute + online softmax + P@V (12 WMMAs).
// Fragment loads are batched ahead of WMMA groups for latency overlap.
// ---------------------------------------------------------------------------
__global__ __launch_bounds__(128) void attn_flash_wmma(
    const _Float16* __restrict__ qw, const _Float16* __restrict__ qr,
    const _Float16* __restrict__ kmat, const _Float16* __restrict__ vT,
    const _Float16* __restrict__ rel, _Float16* __restrict__ attn16) {
  __shared__ _Float16 pbuf[4][16 * 32];       // per-wave P transpose buffer (C->A layout)
  const int w = threadIdx.x >> 5, lane = threadIdx.x & 31;
  const int lm = lane & 15, hi = lane >> 4;
  const int bh = blockIdx.y;                  // b*H + h
  const int b = bh >> 3, h = bh & 7;
  const int tsub = blockIdx.x * 64 + w * 16;

  const _Float16* qwp = qw + (size_t)bh * Tc * QKc;
  const _Float16* qrp = qr + (size_t)bh * Tc * QKc;
  const _Float16* kp  = kmat + (size_t)bh * Tc * QKc;
  const _Float16* vp  = vT + (size_t)bh * Vc * Tc;
  const _Float16* rp  = rel + (size_t)h * Rp * QKc;

  v16h qwA[2], qrA[2];
#pragma unroll
  for (int kc = 0; kc < 2; ++kc) {
    qwA[kc] = load_fragA(qwp + (size_t)(tsub + lm) * QKc + kc * 32, hi);
    qrA[kc] = load_fragA(qrp + (size_t)(tsub + lm) * QKc + kc * 32, hi);
  }

  v8f O[12];
#pragma unroll
  for (int j = 0; j < 12; ++j) O[j] = vzero8();
  float mrow[8], lrow[8];
#pragma unroll
  for (int g = 0; g < 8; ++g) { mrow[g] = -1e30f; lrow[g] = 0.f; }

  for (int s0 = 0; s0 < Tc; s0 += 32) {
    v8f pc0 = vzero8(), pc1 = vzero8();
    v8f band0 = vzero8(), band1 = vzero8(), band2 = vzero8();
    const int rbase = s0 - tsub + (Tc - 16);  // r = rbase + (15 + n - m), always in [0, Rp-1]

    // Prefetch next chunk's streamed operands (global_prefetch_b8).
    if (s0 + 32 < Tc) {
      __builtin_prefetch(kp + (size_t)(s0 + 32 + lane) * QKc, 0, 0);
      __builtin_prefetch(rp + (size_t)(rbase + 32 + lane) * QKc, 0, 0);
    }

#pragma unroll
    for (int kc = 0; kc < 2; ++kc) {
      // Batch the 5 B fragments (10 b128 loads) ahead of the 5 WMMAs.
      const v16h bK0 = load_fragB(kp + (size_t)(s0 + lm) * QKc + kc * 32, hi);
      const v16h bK1 = load_fragB(kp + (size_t)(s0 + 16 + lm) * QKc + kc * 32, hi);
      const v16h bR0 = load_fragB(rp + (size_t)(rbase + lm) * QKc + kc * 32, hi);
      const v16h bR1 = load_fragB(rp + (size_t)(rbase + 16 + lm) * QKc + kc * 32, hi);
      const v16h bR2 = load_fragB(rp + (size_t)(rbase + 32 + lm) * QKc + kc * 32, hi);
      pc0   = wmma_f16(qwA[kc], bK0, pc0);
      pc1   = wmma_f16(qwA[kc], bK1, pc1);
      band0 = wmma_f16(qrA[kc], bR0, band0);
      band1 = wmma_f16(qrA[kc], bR1, band1);
      band2 = wmma_f16(qrA[kc], bR2, band2);
    }
    // Skewed-diagonal extraction: out(m, n) needs band flat-col (n + 15 - m).
    const int lanebase = hi * 16;
#pragma unroll
    for (int g = 0; g < 8; ++g) {
      const int m = g + 8 * hi;
      const int cp = lm + 15 - m;             // [0, 30]
      const int src = lanebase + (cp & 15);   // (cp & 15) == cp - 16 when cp >= 16
      const float e0 = __shfl(band0[g], src, 32);
      const float e1 = __shfl(band1[g], src, 32);
      const float e2 = __shfl(band2[g], src, 32);
      pc0[g] += (cp < 16) ? e0 : e1;
      pc1[g] += (cp < 16) ? e1 : e2;
    }
    // Online softmax: per-row stats replicated across the 16 lanes of each half.
    float alphav[8];
#pragma unroll
    for (int g = 0; g < 8; ++g) {
      float v0 = pc0[g], v1 = pc1[g];
      float mx = fmaxf(v0, v1);
#pragma unroll
      for (int d = 1; d < 16; d <<= 1) mx = fmaxf(mx, __shfl_xor(mx, d, 32));
      const float mnew = fmaxf(mrow[g], mx);
      const float alpha = expf(mrow[g] - mnew);
      mrow[g] = mnew;
      v0 = expf(v0 - mnew);
      v1 = expf(v1 - mnew);
      float rs = v0 + v1;
#pragma unroll
      for (int d = 1; d < 16; d <<= 1) rs += __shfl_xor(rs, d, 32);
      lrow[g] = lrow[g] * alpha + rs;
      alphav[g] = alpha;
      pc0[g] = v0;
      pc1[g] = v1;
    }
#pragma unroll
    for (int j = 0; j < 12; ++j)
#pragma unroll
      for (int g = 0; g < 8; ++g) O[j][g] *= alphav[g];
    // P (16x32, C layout) -> f16 A fragment via per-wave LDS roundtrip.
#pragma unroll
    for (int g = 0; g < 8; ++g) {
      const int m = g + 8 * hi;
      pbuf[w][m * 32 + lm]      = (_Float16)pc0[g];
      pbuf[w][m * 32 + 16 + lm] = (_Float16)pc1[g];
    }
    asm volatile("s_wait_dscnt 0" ::: "memory");
    const v16h pA = load_fragA(&pbuf[w][lm * 32], hi);
    // O(16x192) += P(16x32) @ V(32x192): B fragments from vT rows (contiguous in s).
    // Batched in groups of 4 to overlap loads with WMMAs at bounded VGPR cost.
#pragma unroll
    for (int j0 = 0; j0 < 12; j0 += 4) {
      v16h bV[4];
#pragma unroll
      for (int u = 0; u < 4; ++u)
        bV[u] = load_fragB(vp + (size_t)(16 * (j0 + u) + lm) * Tc + s0, hi);
#pragma unroll
      for (int u = 0; u < 4; ++u)
        O[j0 + u] = wmma_f16(pA, bV[u], O[j0 + u]);
    }
  }
  // Finalize: divide by row sums, write attention output [B, T, H*V] in f16.
#pragma unroll
  for (int g = 0; g < 8; ++g) lrow[g] = 1.f / lrow[g];
#pragma unroll
  for (int j = 0; j < 12; ++j) {
#pragma unroll
    for (int g = 0; g < 8; ++g) {
      const int t = tsub + g + 8 * hi;
      const int vcol = 16 * j + lm;
      attn16[((size_t)(b * Tc + t)) * (Hc * Vc) + h * Vc + vcol] =
          (_Float16)(O[j][g] * lrow[g]);
    }
  }
}

// ---------------------------------------------------------------------------
// Host orchestration
// ---------------------------------------------------------------------------
extern "C" void kernel_launch(void* const* d_in, const int* in_sizes, int n_in,
                              void* d_out, int out_size, void* d_ws, size_t ws_size,
                              hipStream_t stream) {
  (void)in_sizes; (void)n_in; (void)out_size; (void)ws_size;
  const float* x     = (const float*)d_in[0];
  const float* Q_w   = (const float*)d_in[1];
  const float* K_w   = (const float*)d_in[2];
  const float* V_w   = (const float*)d_in[3];
  const float* out_w = (const float*)d_in[4];
  const float* out_b = (const float*)d_in[5];
  const float* rel_w = (const float*)d_in[6];
  const float* r_wb  = (const float*)d_in[7];
  const float* r_rb  = (const float*)d_in[8];

  // Workspace bump allocator (halves)
  _Float16* ws = (_Float16*)d_ws;
  size_t off = 0;
  auto alloc = [&](size_t nhalves) { _Float16* p = ws + off; off += (nhalves + 127) & ~size_t(127); return p; };
  const size_t M = (size_t)Bc * Tc;            // 3072
  _Float16* x16   = alloc(M * Cc);             // [3072,1536]
  _Float16* wtq   = alloc((size_t)Hc * QKc * Cc);       // [512,1536]
  _Float16* wtk   = alloc((size_t)Hc * QKc * Cc);
  _Float16* wtv   = alloc((size_t)Hc * Vc * Cc);        // [1536,1536]
  _Float16* wto   = alloc((size_t)Hc * Vc * Cc);
  _Float16* wtr   = alloc((size_t)Hc * QKc * Fc);       // [512,192]
  _Float16* pf16  = alloc((size_t)Rp * Fc);             // [3072,192]
  _Float16* qw16  = alloc((size_t)Bc * Hc * Tc * QKc);  // [B,H,T,QK]
  _Float16* qr16  = alloc((size_t)Bc * Hc * Tc * QKc);
  _Float16* k16   = alloc((size_t)Bc * Hc * Tc * QKc);
  _Float16* vT16  = alloc((size_t)Bc * Hc * Vc * Tc);   // [B,H,V,T]
  _Float16* rel16 = alloc((size_t)Hc * Rp * QKc);       // [H,3072,64]
  _Float16* at16  = alloc(M * (size_t)(Hc * Vc));       // [3072,1536]

  // 1) precision conversion (+ weight transposes to [N,K])
  cvt_f32_to_f16<<<(int)((M * Cc + 255) / 256), 256, 0, stream>>>(x, x16, (int)(M * Cc));
  cvt_transpose_f16<<<(Cc * Hc * QKc + 255) / 256, 256, 0, stream>>>(Q_w, wtq, Cc, Hc * QKc);
  cvt_transpose_f16<<<(Cc * Hc * QKc + 255) / 256, 256, 0, stream>>>(K_w, wtk, Cc, Hc * QKc);
  cvt_transpose_f16<<<(Cc * Hc * Vc + 255) / 256, 256, 0, stream>>>(V_w, wtv, Cc, Hc * Vc);
  cvt_transpose_f16<<<(Cc * Hc * Vc + 255) / 256, 256, 0, stream>>>(out_w, wto, Cc, Hc * Vc);
  cvt_transpose_f16<<<(Fc * Hc * QKc + 255) / 256, 256, 0, stream>>>(rel_w, wtr, Fc, Hc * QKc);

  // 2) positional features (f16, padded to 3072 rows)
  posfeat_kernel<<<Rp / 4, 128, 0, stream>>>(pf16);

  // 3) projections via WMMA GEMMs with layout/bias epilogues
  gemm_f16_wmma<1><<<dim3(8, 48), 128, 0, stream>>>(x16, wtq, (int)M, Hc * QKc, Cc,
                                                    nullptr, qw16, qr16, r_wb, r_rb);
  gemm_f16_wmma<2><<<dim3(8, 48), 128, 0, stream>>>(x16, wtk, (int)M, Hc * QKc, Cc,
                                                    nullptr, k16, nullptr, nullptr, nullptr);
  gemm_f16_wmma<3><<<dim3(24, 48), 128, 0, stream>>>(x16, wtv, (int)M, Hc * Vc, Cc,
                                                     nullptr, vT16, nullptr, nullptr, nullptr);
  gemm_f16_wmma<4><<<dim3(8, 48), 128, 0, stream>>>(pf16, wtr, Rp, Hc * QKc, Fc,
                                                    nullptr, rel16, nullptr, nullptr, nullptr);

  // 4) fused flash attention with folded relative shift
  attn_flash_wmma<<<dim3(Tc / 64, Bc * Hc), 128, 0, stream>>>(qw16, qr16, k16, vT16, rel16, at16);

  // 5) output projection + bias -> d_out (f32)
  gemm_f16_wmma<0><<<dim3(24, 48), 128, 0, stream>>>(at16, wto, (int)M, Hc * Vc, Cc,
                                                     (float*)d_out, nullptr, nullptr, out_b, nullptr);
}
